// MultiHeadSelfAttention_86895778332899
// MI455X (gfx1250) — matrix operand order repaired and checked by reading, not compiled, not run on hardware
//
#include <hip/hip_runtime.h>
#include <hip/hip_bf16.h>

// ---------------- problem constants ----------------
#define DIMX 512
#define HEADS 8
#define DHEAD 64
#define INNER 512
#define NB 8
#define NN 1024
#define NROWS (NB * NN) // 8192

typedef __bf16 bf16_t;
typedef __attribute__((ext_vector_type(16))) __bf16 v16bf;
typedef __attribute__((ext_vector_type(8)))  __bf16 v8bf;
typedef __attribute__((ext_vector_type(8)))  float  v8f;

// ---- CDNA5 async global->LDS copy (ASYNCcnt path), with safe fallback ----
#if defined(__has_builtin)
#  if __has_builtin(__builtin_amdgcn_global_load_async_to_lds_b128) && \
      __has_builtin(__builtin_amdgcn_s_wait_asynccnt)
#    define USE_ASYNC_LDS 1
#  endif
#endif
#ifndef USE_ASYNC_LDS
#  define USE_ASYNC_LDS 0
#endif

#if USE_ASYNC_LDS
typedef int v4i_gcc __attribute__((vector_size(4 * sizeof(int))));
typedef __attribute__((address_space(1))) v4i_gcc glb_v4i;
typedef __attribute__((address_space(3))) v4i_gcc lds_v4i;
static __device__ __forceinline__ void async_cp_b128(const bf16_t* g, bf16_t* l) {
    __builtin_amdgcn_global_load_async_to_lds_b128((glb_v4i*)g, (lds_v4i*)l, 0, 0);
}
#endif

static __device__ __forceinline__ bf16_t f2bf(float f) {
    unsigned u = __float_as_uint(f);
    unsigned r = (u + 0x7FFFu + ((u >> 16) & 1u)) >> 16; // RNE
    unsigned short s = (unsigned short)r;
    return __builtin_bit_cast(bf16_t, s);
}
static __device__ __forceinline__ float bf2f(bf16_t b) {
    unsigned short s = __builtin_bit_cast(unsigned short, b);
    return __uint_as_float(((unsigned)s) << 16);
}

// A-tile 16(M)x32(K) bf16 from row-major [ld] source.
// Lane l<16: row=l, K=0..7 / 16..23 ; lane l>=16: row=l-16, K=8..15 / 24..31.
static __device__ __forceinline__ v16bf load_a16x32(const bf16_t* p, int ld, int lane) {
    int r  = lane & 15;
    int kb = (lane >> 4) << 3; // 0 or 8
    v8bf lo = *(const v8bf*)(p + (size_t)r * ld + kb);
    v8bf hi = *(const v8bf*)(p + (size_t)r * ld + kb + 16);
    v16bf a;
#pragma unroll
    for (int i = 0; i < 8; i++) { a[i] = lo[i]; a[i + 8] = hi[i]; }
    return a;
}

// B-tile 32(K)x16(N) bf16 where memory holds the transpose: pT[n][k], stride ld.
// Lanes 0-15 hold K=0..7 / 16..23 for col n=lane; lanes 16-31 hold K=8..15 / 24..31.
static __device__ __forceinline__ v16bf load_b32x16(const bf16_t* pT, int ld, int lane) {
    int n  = lane & 15;
    int kh = (lane >> 4) << 3; // 0 or 8
    v8bf lo = *(const v8bf*)(pT + (size_t)n * ld + kh);
    v8bf hi = *(const v8bf*)(pT + (size_t)n * ld + kh + 16);
    v16bf b;
#pragma unroll
    for (int i = 0; i < 8; i++) { b[i] = lo[i]; b[i + 8] = hi[i]; }
    return b;
}

static __device__ __forceinline__ v8f wmma_bf16(v16bf a, v16bf b, v8f c) {
    return __builtin_amdgcn_wmma_f32_16x16x32_bf16(false, a, false, b, (short)0, c, false, false);
}

// ---------------- kernel 0: weight convert + transpose (32x32 LDS tiles) ----------------
// grid: 768 (wqkv 16x48 tiles) + 256 (wout 16x16 tiles) = 1024 blocks of 256
__global__ void k_prep_weights(const float* __restrict__ wqkv, const float* __restrict__ wout,
                               bf16_t* __restrict__ wqkvT, bf16_t* __restrict__ woutT) {
    __shared__ float tile[32][33];
    int blk = blockIdx.x;
    const float* src; bf16_t* dst; int srcCols, dstCols, tr, tc;
    if (blk < 768) {           // wqkv: 512 x 1536 -> wqkvT: 1536 x 512
        src = wqkv; dst = wqkvT; srcCols = 3 * INNER; dstCols = DIMX;
        tr = blk / 48; tc = blk % 48;
    } else {                   // wout: 512 x 512 -> woutT: 512 x 512
        int u = blk - 768;
        src = wout; dst = woutT; srcCols = DIMX; dstCols = INNER;
        tr = u / 16; tc = u % 16;
    }
    int tx  = threadIdx.x & 31;
    int ty0 = threadIdx.x >> 5; // 0..7
#pragma unroll
    for (int i = 0; i < 4; i++) {
        int r = ty0 + i * 8;
        tile[r][tx] = src[(size_t)(tr * 32 + r) * srcCols + tc * 32 + tx];
    }
    __syncthreads();
#pragma unroll
    for (int i = 0; i < 4; i++) {
        int r = ty0 + i * 8;
        dst[(size_t)(tc * 32 + r) * dstCols + tr * 32 + tx] = f2bf(tile[tx][r]);
    }
}

// ---------------- kernel 1: precompute relative-position bias (H,N,N) bf16 ----------------
__global__ void k_bias(const int* __restrict__ rel_idx, const float* __restrict__ rel_emb,
                       bf16_t* __restrict__ bias) {
    int t = blockIdx.x * blockDim.x + threadIdx.x; // < NN*NN
    int idx = rel_idx[t];
    const float* e = rel_emb + (size_t)idx * HEADS;
#pragma unroll
    for (int h = 0; h < HEADS; h++)
        bias[(size_t)h * NN * NN + t] = f2bf(e[h]);
}

// ---------------- kernel 2: LayerNorm + cast to bf16 (one wave per 512-row) ----------------
__global__ void k_layernorm(const float* __restrict__ x, const float* __restrict__ gamma,
                            const float* __restrict__ beta, bf16_t* __restrict__ xn) {
    int lane = threadIdx.x & 31;
    int wid  = threadIdx.x >> 5;
    int row  = blockIdx.x * 8 + wid;
    float vals[16];
    const float4* p4 = (const float4*)(x + (size_t)row * DIMX + lane * 16);
#pragma unroll
    for (int i = 0; i < 4; i++) {
        float4 v = p4[i];
        vals[4 * i] = v.x; vals[4 * i + 1] = v.y; vals[4 * i + 2] = v.z; vals[4 * i + 3] = v.w;
    }
    float s = 0.f, ss = 0.f;
#pragma unroll
    for (int i = 0; i < 16; i++) { s += vals[i]; ss += vals[i] * vals[i]; }
#pragma unroll
    for (int m = 1; m < 32; m <<= 1) { s += __shfl_xor(s, m, 32); ss += __shfl_xor(ss, m, 32); }
    float mean = s * (1.f / DIMX);
    float var  = ss * (1.f / DIMX) - mean * mean;
    float rstd = rsqrtf(var + 1e-5f);
    bf16_t* o = xn + (size_t)row * DIMX + lane * 16;
#pragma unroll
    for (int i = 0; i < 16; i++) {
        int c = lane * 16 + i;
        o[i] = f2bf((vals[i] - mean) * rstd * gamma[c] + beta[c]);
    }
}

// ---------------- kernel 3: QKV GEMM (8192x512 @ 512x1536), double-buffered ----------------
__global__ __launch_bounds__(128) void k_qkv_gemm(const bf16_t* __restrict__ xn,
                                                  const bf16_t* __restrict__ wqkvT,
                                                  bf16_t* __restrict__ q, bf16_t* __restrict__ kmat,
                                                  bf16_t* __restrict__ vT) {
    int lane = threadIdx.x & 31;
    int w = blockIdx.x * 4 + (threadIdx.x >> 5);
    int rowTile = w / 24;
    int cg      = w % 24; // 64-col group of 1536
    const bf16_t* arow = xn + (size_t)rowTile * 16 * DIMX;
    const bf16_t* wp   = wqkvT + (size_t)cg * 64 * DIMX;
    v8f acc[4] = {};
    v16bf a  = load_a16x32(arow, DIMX, lane);
    v16bf b0 = load_b32x16(wp + (size_t)0  * DIMX, DIMX, lane);
    v16bf b1 = load_b32x16(wp + (size_t)16 * DIMX, DIMX, lane);
    v16bf b2 = load_b32x16(wp + (size_t)32 * DIMX, DIMX, lane);
    v16bf b3 = load_b32x16(wp + (size_t)48 * DIMX, DIMX, lane);
    for (int kk = 32; kk < DIMX; kk += 32) {
        v16bf an  = load_a16x32(arow + kk, DIMX, lane);
        v16bf bn0 = load_b32x16(wp + (size_t)0  * DIMX + kk, DIMX, lane);
        v16bf bn1 = load_b32x16(wp + (size_t)16 * DIMX + kk, DIMX, lane);
        v16bf bn2 = load_b32x16(wp + (size_t)32 * DIMX + kk, DIMX, lane);
        v16bf bn3 = load_b32x16(wp + (size_t)48 * DIMX + kk, DIMX, lane);
        acc[0] = wmma_bf16(a, b0, acc[0]);
        acc[1] = wmma_bf16(a, b1, acc[1]);
        acc[2] = wmma_bf16(a, b2, acc[2]);
        acc[3] = wmma_bf16(a, b3, acc[3]);
        a = an; b0 = bn0; b1 = bn1; b2 = bn2; b3 = bn3;
    }
    acc[0] = wmma_bf16(a, b0, acc[0]);
    acc[1] = wmma_bf16(a, b1, acc[1]);
    acc[2] = wmma_bf16(a, b2, acc[2]);
    acc[3] = wmma_bf16(a, b3, acc[3]);

    int c0 = cg * 64;
    int which = c0 >> 9;           // 0=q 1=k 2=v
    int head  = (c0 & 511) >> 6;
    int row0  = rowTile * 16;
    int b_    = row0 >> 10;
    int n0    = row0 & 1023;
    int n  = lane & 15;
    int mb = (lane >> 4) << 3;
    if (which < 2) {
        bf16_t* qk = (which == 0) ? q : kmat;
#pragma unroll
        for (int g = 0; g < 4; g++) {
            int d = g * 16 + n;
#pragma unroll
            for (int r = 0; r < 8; r++) {
                int nr = n0 + mb + r;
                qk[((size_t)(b_ * HEADS + head) * NN + nr) * DHEAD + d] = f2bf(acc[g][r]);
            }
        }
    } else {
        // vT[bh][d][n] : 8 consecutive n per lane -> one b128 store per tile
#pragma unroll
        for (int g = 0; g < 4; g++) {
            int d = g * 16 + n;
            v8bf pk;
#pragma unroll
            for (int r = 0; r < 8; r++) pk[r] = f2bf(acc[g][r]);
            *(v8bf*)(vT + ((size_t)(b_ * HEADS + head) * DHEAD + d) * NN + n0 + mb) = pk;
        }
    }
}

// ---------------- kernel 4: flash attention ----------------
// 1 block = 4 waves sharing (batch, head); each wave owns a 16-query tile.
// K/V chunks staged in LDS: async DMA double-buffer on CDNA5, register prefetch fallback.
__global__ __launch_bounds__(128) void k_attention(const bf16_t* __restrict__ q,
                                                   const bf16_t* __restrict__ kmat,
                                                   const bf16_t* __restrict__ vT,
                                                   const bf16_t* __restrict__ bias,
                                                   const unsigned char* __restrict__ mask,
                                                   bf16_t* __restrict__ attn_out) {
#if USE_ASYNC_LDS
    __shared__ alignas(16) bf16_t ldsK[2][32 * DHEAD];  // 2 x 4KB
    __shared__ alignas(16) bf16_t ldsV[2][DHEAD * 32];  // 2 x 4KB
#else
    __shared__ alignas(16) bf16_t ldsK[1][32 * DHEAD];
    __shared__ alignas(16) bf16_t ldsV[1][DHEAD * 32];
#endif
    __shared__ alignas(16) bf16_t ptile[4][16][32];     // per-wave P transpose buffer
    int tid  = threadIdx.x;
    int lane = tid & 31;
    int wid  = tid >> 5;
    int bh = blockIdx.x >> 4;                 // (b*HEADS + h), 16 blocks per head
    int b_ = bh >> 3;
    int h  = bh & 7;
    int qt = (blockIdx.x & 15) * 4 + wid;

    const bf16_t* qp = q    + ((size_t)bh * NN + qt * 16) * DHEAD;
    const bf16_t* kp = kmat + (size_t)bh * NN * DHEAD;
    const bf16_t* vp = vT   + (size_t)bh * DHEAD * NN;
    const bf16_t* bp = bias + ((size_t)h * NN + qt * 16) * NN;
    const unsigned char* mp = mask + ((size_t)b_ * NN + qt * 16) * NN;

    v16bf aq0 = load_a16x32(qp, DHEAD, lane);
    v16bf aq1 = load_a16x32(qp + 32, DHEAD, lane);

    int n  = lane & 15;
    int mb = (lane >> 4) << 3;

    float m[8], l[8];
    v8f o[4] = {};
#pragma unroll
    for (int r = 0; r < 8; r++) { m[r] = -3.0e38f; l[r] = 0.f; }
    const float scale = 0.125f; // 1/sqrt(64)

    // cooperative addressing: K chunk = 4KB contiguous; V chunk = 64 rows x 64B
    int vd1 = tid >> 2,          vq1 = tid & 3;
    int vd2 = (tid + 128) >> 2,  vq2 = (tid + 128) & 3;

#if USE_ASYNC_LDS
    // DMA chunk 0 into buffer 0 (no VGPR involvement, tracked by ASYNCcnt)
    async_cp_b128(kp + tid * 8,                         &ldsK[0][tid * 8]);
    async_cp_b128(kp + tid * 8 + 1024,                  &ldsK[0][tid * 8 + 1024]);
    async_cp_b128(vp + (size_t)vd1 * NN + vq1 * 8,      &ldsV[0][vd1 * 32 + vq1 * 8]);
    async_cp_b128(vp + (size_t)vd2 * NN + vq2 * 8,      &ldsV[0][vd2 * 32 + vq2 * 8]);
    __builtin_amdgcn_s_wait_asynccnt(0);
    __syncthreads();
#else
    // register prefetch of chunk 0
    v8bf kr0 = *(const v8bf*)(kp + tid * 8);
    v8bf kr1 = *(const v8bf*)(kp + tid * 8 + 1024);
    v8bf vr0 = *(const v8bf*)(vp + (size_t)vd1 * NN + vq1 * 8);
    v8bf vr1 = *(const v8bf*)(vp + (size_t)vd2 * NN + vq2 * 8);
#endif

    for (int jt = 0; jt < NN / 32; jt++) {
        int key0 = jt * 32;
#if USE_ASYNC_LDS
        const bf16_t* kc = ldsK[jt & 1];
        const bf16_t* vc = ldsV[jt & 1];
        if (jt + 1 < NN / 32) {  // DMA next chunk into the other buffer, overlapping compute
            int keyn = key0 + 32, nb = (jt + 1) & 1;
            async_cp_b128(kp + (size_t)keyn * DHEAD + tid * 8,        &ldsK[nb][tid * 8]);
            async_cp_b128(kp + (size_t)keyn * DHEAD + tid * 8 + 1024, &ldsK[nb][tid * 8 + 1024]);
            async_cp_b128(vp + (size_t)vd1 * NN + keyn + vq1 * 8,     &ldsV[nb][vd1 * 32 + vq1 * 8]);
            async_cp_b128(vp + (size_t)vd2 * NN + keyn + vq2 * 8,     &ldsV[nb][vd2 * 32 + vq2 * 8]);
        }
#else
        *(v8bf*)(ldsK[0] + tid * 8)            = kr0;
        *(v8bf*)(ldsK[0] + tid * 8 + 1024)     = kr1;
        *(v8bf*)(ldsV[0] + vd1 * 32 + vq1 * 8) = vr0;
        *(v8bf*)(ldsV[0] + vd2 * 32 + vq2 * 8) = vr1;
        __syncthreads();
        if (jt + 1 < NN / 32) {  // register prefetch of next chunk
            int keyn = key0 + 32;
            kr0 = *(const v8bf*)(kp + (size_t)keyn * DHEAD + tid * 8);
            kr1 = *(const v8bf*)(kp + (size_t)keyn * DHEAD + tid * 8 + 1024);
            vr0 = *(const v8bf*)(vp + (size_t)vd1 * NN + keyn + vq1 * 8);
            vr1 = *(const v8bf*)(vp + (size_t)vd2 * NN + keyn + vq2 * 8);
        }
        const bf16_t* kc = ldsK[0];
        const bf16_t* vc = ldsV[0];
#endif
        // S = Q K^T from LDS (two 16x16 key tiles, d accumulated in two wmma each)
        v8f s0 = {}, s1 = {};
        {
            v16bf kb0 = load_b32x16(kc,               DHEAD, lane);
            v16bf kb1 = load_b32x16(kc + 32,          DHEAD, lane);
            s0 = wmma_bf16(aq0, kb0, s0);
            s0 = wmma_bf16(aq1, kb1, s0);
            v16bf kc0 = load_b32x16(kc + 16 * DHEAD,      DHEAD, lane);
            v16bf kc1 = load_b32x16(kc + 16 * DHEAD + 32, DHEAD, lane);
            s1 = wmma_bf16(aq0, kc0, s1);
            s1 = wmma_bf16(aq1, kc1, s1);
        }
        // scale + bias + mask (C layout: col = key0 [+16] + n, rows mb..mb+7)
#pragma unroll
        for (int r = 0; r < 8; r++) {
            int rowq = mb + r;
            int c0i = key0 + n, c1i = key0 + 16 + n;
            float v0 = s0[r] * scale + bf2f(bp[(size_t)rowq * NN + c0i]);
            float v1 = s1[r] * scale + bf2f(bp[(size_t)rowq * NN + c1i]);
            if (!mp[(size_t)rowq * NN + c0i]) v0 = -3.0e38f;
            if (!mp[(size_t)rowq * NN + c1i]) v1 = -3.0e38f;
            s0[r] = v0; s1[r] = v1;
        }
        // online softmax: row reductions across 16 lanes of each half-wave
        float alpha[8];
#pragma unroll
        for (int r = 0; r < 8; r++) {
            float mx = fmaxf(s0[r], s1[r]);
#pragma unroll
            for (int msk = 1; msk < 16; msk <<= 1) mx = fmaxf(mx, __shfl_xor(mx, msk, 32));
            float nm = fmaxf(m[r], mx);
            alpha[r] = __expf(m[r] - nm);
            m[r] = nm;
            float p0 = __expf(s0[r] - nm);
            float p1 = __expf(s1[r] - nm);
            s0[r] = p0; s1[r] = p1;
            float rs = p0 + p1;
#pragma unroll
            for (int msk = 1; msk < 16; msk <<= 1) rs += __shfl_xor(rs, msk, 32);
            l[r] = l[r] * alpha[r] + rs;
        }
#pragma unroll
        for (int g = 0; g < 4; g++)
#pragma unroll
            for (int r = 0; r < 8; r++) o[g][r] *= alpha[r];

        // transpose P (C layout -> A layout) through LDS, converting to bf16
        __syncthreads();
#pragma unroll
        for (int r = 0; r < 8; r++) {
            ptile[wid][mb + r][n]      = f2bf(s0[r]);
            ptile[wid][mb + r][16 + n] = f2bf(s1[r]);
        }
        __syncthreads();
        v16bf pa = load_a16x32(&ptile[wid][0][0], 32, lane);

        // O += P @ V from LDS (vT chunk: rows = d, cols = key)
#pragma unroll
        for (int g = 0; g < 4; g++) {
            v16bf vb = load_b32x16(vc + (size_t)(g * 16) * 32, 32, lane);
            o[g] = wmma_bf16(pa, vb, o[g]);
        }
#if USE_ASYNC_LDS
        if (jt + 1 < NN / 32) __builtin_amdgcn_s_wait_asynccnt(0);
        __syncthreads(); // next buffer filled + everyone done reading current
#else
        __syncthreads(); // protect ldsK/ldsV before next iteration's stores
#endif
    }
    // normalize and write (B, N, H*64) bf16
#pragma unroll
    for (int g = 0; g < 4; g++) {
        int d = h * DHEAD + g * 16 + n;
#pragma unroll
        for (int r = 0; r < 8; r++) {
            float inv = (l[r] > 0.f) ? (1.f / l[r]) : 0.f;
            attn_out[((size_t)b_ * NN + qt * 16 + mb + r) * INNER + d] = f2bf(o[g][r] * inv);
        }
    }
}

// ---------------- kernel 5: output projection + bias + residual, double-buffered ----------------
__global__ __launch_bounds__(128) void k_out_proj(const bf16_t* __restrict__ attn_out,
                                                  const bf16_t* __restrict__ woutT,
                                                  const float* __restrict__ b_out,
                                                  const float* __restrict__ x,
                                                  float* __restrict__ out) {
    int lane = threadIdx.x & 31;
    int w = blockIdx.x * 4 + (threadIdx.x >> 5);
    int rowTile = w >> 3;
    int cg      = w & 7;
    const bf16_t* arow = attn_out + (size_t)rowTile * 16 * INNER;
    const bf16_t* wp   = woutT + (size_t)cg * 64 * INNER;
    v8f acc[4] = {};
    v16bf a  = load_a16x32(arow, INNER, lane);
    v16bf b0 = load_b32x16(wp + (size_t)0  * INNER, INNER, lane);
    v16bf b1 = load_b32x16(wp + (size_t)16 * INNER, INNER, lane);
    v16bf b2 = load_b32x16(wp + (size_t)32 * INNER, INNER, lane);
    v16bf b3 = load_b32x16(wp + (size_t)48 * INNER, INNER, lane);
    for (int kk = 32; kk < INNER; kk += 32) {
        v16bf an  = load_a16x32(arow + kk, INNER, lane);
        v16bf bn0 = load_b32x16(wp + (size_t)0  * INNER + kk, INNER, lane);
        v16bf bn1 = load_b32x16(wp + (size_t)16 * INNER + kk, INNER, lane);
        v16bf bn2 = load_b32x16(wp + (size_t)32 * INNER + kk, INNER, lane);
        v16bf bn3 = load_b32x16(wp + (size_t)48 * INNER + kk, INNER, lane);
        acc[0] = wmma_bf16(a, b0, acc[0]);
        acc[1] = wmma_bf16(a, b1, acc[1]);
        acc[2] = wmma_bf16(a, b2, acc[2]);
        acc[3] = wmma_bf16(a, b3, acc[3]);
        a = an; b0 = bn0; b1 = bn1; b2 = bn2; b3 = bn3;
    }
    acc[0] = wmma_bf16(a, b0, acc[0]);
    acc[1] = wmma_bf16(a, b1, acc[1]);
    acc[2] = wmma_bf16(a, b2, acc[2]);
    acc[3] = wmma_bf16(a, b3, acc[3]);

    int n = lane & 15, mb = (lane >> 4) << 3;
#pragma unroll
    for (int g = 0; g < 4; g++) {
        int c = cg * 64 + g * 16 + n;
        float bo = b_out[c];
#pragma unroll
        for (int r = 0; r < 8; r++) {
            size_t idx = ((size_t)rowTile * 16 + mb + r) * DIMX + c;
            out[idx] = acc[g][r] + bo + x[idx];
        }
    }
}

// ---------------- host launch ----------------
extern "C" void kernel_launch(void* const* d_in, const int* in_sizes, int n_in,
                              void* d_out, int out_size, void* d_ws, size_t ws_size,
                              hipStream_t stream) {
    const float* x            = (const float*)d_in[0];
    const unsigned char* mask = (const unsigned char*)d_in[1];
    const float* gamma        = (const float*)d_in[2];
    const float* beta         = (const float*)d_in[3];
    const float* w_qkv        = (const float*)d_in[4];
    const float* w_out        = (const float*)d_in[5];
    const float* b_out        = (const float*)d_in[6];
    const float* rel_emb      = (const float*)d_in[7];
    const int*   rel_idx      = (const int*)d_in[8];
    float* out = (float*)d_out;

    char* ws = (char*)d_ws;
    size_t off = 0;
    auto alloc = [&](size_t bytes) -> void* {
        void* p = ws + off;
        off += (bytes + 255) & ~(size_t)255;
        return p;
    };
    bf16_t* xn    = (bf16_t*)alloc((size_t)NROWS * DIMX * 2);            // 8 MiB
    bf16_t* wqkvT = (bf16_t*)alloc((size_t)3 * INNER * DIMX * 2);        // 1.5 MiB
    bf16_t* woutT = (bf16_t*)alloc((size_t)INNER * DIMX * 2);            // 0.5 MiB
    bf16_t* qb    = (bf16_t*)alloc((size_t)NB * HEADS * NN * DHEAD * 2); // 8 MiB
    bf16_t* kb    = (bf16_t*)alloc((size_t)NB * HEADS * NN * DHEAD * 2); // 8 MiB
    bf16_t* vtb   = (bf16_t*)alloc((size_t)NB * HEADS * DHEAD * NN * 2); // 8 MiB
    bf16_t* biasb = (bf16_t*)alloc((size_t)HEADS * NN * NN * 2);         // 16 MiB
    bf16_t* aout  = (bf16_t*)alloc((size_t)NROWS * INNER * 2);           // 8 MiB
    (void)ws_size; (void)in_sizes; (void)n_in; (void)out_size;

    k_prep_weights<<<1024, 256, 0, stream>>>(w_qkv, w_out, wqkvT, woutT);
    k_bias<<<4096, 256, 0, stream>>>(rel_idx, rel_emb, biasb);
    k_layernorm<<<NROWS / 8, 256, 0, stream>>>(x, gamma, beta, xn);
    k_qkv_gemm<<<(512 * 24) / 4, 128, 0, stream>>>(xn, wqkvT, qb, kb, vtb);
    k_attention<<<(NB * HEADS * (NN / 16)) / 4, 128, 0, stream>>>(qb, kb, vtb, biasb, mask, aout);
    k_out_proj<<<(512 * 8) / 4, 128, 0, stream>>>(aout, woutT, b_out, x, out);
}
